// TemporalAttention_lin_15547781612052
// MI455X (gfx1250) — compile-verified, matrix operand order checked by reading, not run on hardware
//
#include <hip/hip_runtime.h>

typedef __attribute__((ext_vector_type(16))) _Float16     v16h;
typedef __attribute__((ext_vector_type(8)))  float        v8f;
typedef __attribute__((ext_vector_type(4)))  unsigned int v4u;
typedef __attribute__((ext_vector_type(4)))  int          v4i;

#define DEVINL __device__ __forceinline__

constexpr int   CB      = 2;
constexpr int   CC      = 512;
constexpr int   CT      = 16;
constexpr int   CHH     = 32;
constexpr int   CWW     = 32;
constexpr int   CHD     = 64;   // CC / num_heads
constexpr int   SEQ     = CB * CHH * CWW;   // 2048
constexpr int   ROWS    = SEQ * CT;         // 32768
constexpr int   C3      = 3 * CC;           // 1536
constexpr float LNEPS   = 1e-5f;
constexpr float QKSCALE = 0.3535533905932738f; // 64^(-1/4)

// ---------------------------------------------------------------------------
// Async global -> LDS staging (CDNA5 GLOBAL_LOAD_ASYNC_TO_LDS_B128, ASYNCcnt).
// The builtin exists on this toolchain (probe-confirmed by round-2 diagnostic);
// its params are v4i pointers in addrspace(1) (global) and addrspace(3) (LDS).
// ---------------------------------------------------------------------------
#if __has_builtin(__builtin_amdgcn_global_load_async_to_lds_b128)
#define HAVE_ASYNC_LDS 1
#else
#define HAVE_ASYNC_LDS 0
#endif

typedef __attribute__((address_space(1))) v4i* gv4i_ptr;
typedef __attribute__((address_space(3))) v4i* lv4i_ptr;

DEVINL void async_copy_b128(const _Float16* gsrc, _Float16* ldst) {
#if HAVE_ASYNC_LDS
  __builtin_amdgcn_global_load_async_to_lds_b128(
      (gv4i_ptr)gsrc, (lv4i_ptr)ldst, 0, 0);
#else
  *reinterpret_cast<v4u*>(ldst) = *reinterpret_cast<const v4u*>(gsrc);
#endif
}

DEVINL void wait_async_lds() {
#if HAVE_ASYNC_LDS
#if __has_builtin(__builtin_amdgcn_s_wait_asynccnt)
  __builtin_amdgcn_s_wait_asynccnt(0);
#else
  asm volatile("s_wait_asynccnt 0x0" ::: "memory");
#endif
#endif
}

// ---------------------------------------------------------------------------
// WMMA fragment helpers (CDNA5 wave32 layouts, 16-bit A/B 16x32)
// A: lane L -> M = L&15, half = L>>4; VGPR v holds K pair:
//    v<4 : K = 2v + half*8       (bytes 0..15 of [rowp + half*8])
//    v>=4: K = 16+2(v-4)+half*8  (bytes 0..15 of [rowp + half*8 + 16])
// B (K x N): identical indexing with N = L&15 reading row N of the
//            N-major (i.e. transposed) operand.
// ---------------------------------------------------------------------------
union FragU { struct { v4u lo, hi; } u; v16h h; };

DEVINL v16h load_frag(const _Float16* p, int ld) {
  const int lane = threadIdx.x & 31;
  const int mr   = lane & 15;
  const int half = lane >> 4;
  const _Float16* rp = p + mr * ld + half * 8;
  FragU f;
  f.u.lo = *reinterpret_cast<const v4u*>(rp);
  f.u.hi = *reinterpret_cast<const v4u*>(rp + 16);
  return f.h;
}

// B fragment with only K=0..15 valid (K=16..31 zero), operand stored
// N-major with row stride 16 halfwords (LDS-transposed V tile).
DEVINL v16h load_frag_k16(const _Float16* rowp) {
  const int half = (threadIdx.x >> 4) & 1;
  FragU f;
  f.u.lo = *reinterpret_cast<const v4u*>(rowp + half * 8);
  v4u z = {0u, 0u, 0u, 0u};
  f.u.hi = z;
  return f.h;
}

DEVINL v8f wmma_f16(v16h a, v16h b, v8f c) {
  return __builtin_amdgcn_wmma_f32_16x16x32_f16(false, a, false, b,
                                                (short)0, c, false, false);
}

// ---------------------------------------------------------------------------
// K0: f32 -> f16 weight conversion
// ---------------------------------------------------------------------------
__global__ void k_f32_to_f16(const float* __restrict__ src,
                             _Float16* __restrict__ dst, int n) {
  int i = blockIdx.x * blockDim.x + threadIdx.x;
  if (i < n) dst[i] = (_Float16)src[i];
}

// ---------------------------------------------------------------------------
// K1: gather + LayerNorm -> f16 nrm (ROWS x CC), row R = seqB*16 + t,
//     seqB = b*1024 + hi*32 + wi.  Lane = wi -> coalesced x reads.
// ---------------------------------------------------------------------------
__global__ void k_layernorm(const float* __restrict__ x,
                            const float* __restrict__ g,
                            const float* __restrict__ bt,
                            _Float16* __restrict__ nrm) {
  __shared__ float s_sum[8][32];
  __shared__ float s_sq[8][32];
  const int bid = blockIdx.x;            // (b, t, hi)
  const int b  = bid >> 9;
  const int t  = (bid >> 5) & 15;
  const int hi = bid & 31;
  const int wi = threadIdx.x;            // 0..31
  const int cg = threadIdx.y;            // 0..7

  const size_t base = (size_t)b * (CC * CT * CHH * CWW) + (size_t)t * 1024 +
                      (size_t)hi * 32 + wi;
  float s = 0.f, sq = 0.f;
  for (int ci = cg * 64; ci < cg * 64 + 64; ++ci) {
    float v = x[base + (size_t)ci * (CT * 1024)];
    s += v; sq += v * v;
  }
  s_sum[cg][wi] = s;
  s_sq[cg][wi]  = sq;
  __syncthreads();
  float ts = 0.f, tq = 0.f;
#pragma unroll
  for (int i = 0; i < 8; ++i) { ts += s_sum[i][wi]; tq += s_sq[i][wi]; }
  const float mean = ts * (1.0f / 512.0f);
  const float var  = tq * (1.0f / 512.0f) - mean * mean;
  const float rstd = rsqrtf(var + LNEPS);

  const size_t row = ((size_t)(b * 1024 + hi * 32 + wi) * 16 + t);
  for (int ci = cg * 64; ci < cg * 64 + 64; ++ci) {
    float v  = x[base + (size_t)ci * (CT * 1024)];
    float nv = (v - mean) * rstd * g[ci] + bt[ci];
    nrm[row * CC + ci] = (_Float16)nv;
  }
}

// ---------------------------------------------------------------------------
// K2: QKV GEMM  qkv[R][j] = (sum_c nrm[R][c] * Wq[j][c] + b[j]) * scale(j)
//     8 waves/block: 4 row-tiles x 2 col-tiles of 16x16, K-loop 512/32.
//     Weight tile (32 cols x 512 K, 32 KB) staged once via async->LDS.
// ---------------------------------------------------------------------------
__global__ void k_qkv_gemm(const _Float16* __restrict__ nrm,
                           const _Float16* __restrict__ wq,
                           const float* __restrict__ qb,
                           _Float16* __restrict__ qkv) {
  __shared__ _Float16 s_b[32 * CC];      // 32 weight rows (block cols) x K
  const int tid  = threadIdx.x;
  const int w    = tid >> 5;
  const int lane = tid & 31;
  const int n    = lane & 15;
  const int half = lane >> 4;
  const int rt   = w & 3;
  const int ct   = w >> 2;
  const int row0 = blockIdx.y * 64 + rt * 16;
  const int col0 = blockIdx.x * 32 + ct * 16;

  // async stage of the block's weight tile into LDS
  {
    const _Float16* btile = wq + (size_t)(blockIdx.x * 32) * CC;
#pragma unroll
    for (int it = 0; it < 8; ++it) {
      int i = (it * 256 + tid) * 8;      // 2048 x b128 chunks
      async_copy_b128(btile + i, &s_b[i]);
    }
    wait_async_lds();
  }
  __syncthreads();

  const _Float16* abase = nrm + (size_t)row0 * CC;
  const _Float16* bbase = &s_b[(size_t)(ct * 16) * CC];

  v8f acc = {};
#pragma unroll 4
  for (int kk = 0; kk < 16; ++kk) {
    v16h af = load_frag(abase + kk * 32, CC);   // global (streams HBM)
    v16h bf = load_frag(bbase + kk * 32, CC);   // LDS (ds_load_b128)
    acc = wmma_f16(af, bf, acc);
  }

  const int   j  = col0 + n;
  const float bj = qb[j];
  const float sc = (j < 2 * CC) ? QKSCALE : 1.0f; // scale q and k streams
#pragma unroll
  for (int v = 0; v < 8; ++v) {
    const int m = v + 8 * half;
    float val = (acc[v] + bj) * sc;
    qkv[(size_t)(row0 + m) * C3 + j] = (_Float16)val;
  }
}

// ---------------------------------------------------------------------------
// K3: attention. Block = one sequence (seqB), wave = one head.
//     S = q k^T (2 WMMAs) + rel-pos key term (VALU, diagonal table slice),
//     softmax in f32, AV = 4 zero-padded WMMAs + rel-pos value term (VALU).
// ---------------------------------------------------------------------------
__global__ void k_attention(const _Float16* __restrict__ qkv,
                            const float* __restrict__ ktab,
                            const float* __restrict__ vtab,
                            _Float16* __restrict__ aout) {
  __shared__ float    s_kt[31][64];        // k_table[49+i] * scale
  __shared__ float    s_vt[31][64];        // v_table[49+i]
  __shared__ float    s_S[8][16][16];      // scores / attn (f32)
  __shared__ _Float16 s_attnh[8][16][32];  // attn f16, K padded to 32
  __shared__ _Float16 s_vT[8][64][16];     // V transposed: [c][k]

  const int tid  = threadIdx.x;
  const int w    = tid >> 5;
  const int lane = tid & 31;
  const int n    = lane & 15;
  const int half = lane >> 4;
  const int seqB = blockIdx.x;
  const int hd   = w;

  // --- preload: rel-pos table slices (rows 49..79), zero attn pad, V^T ---
  for (int i = tid; i < 31 * 64; i += 256) {
    int r = i >> 6, c = i & 63;
    s_kt[r][c] = ktab[(49 + r) * 64 + c] * QKSCALE;
    s_vt[r][c] = vtab[(49 + r) * 64 + c];
  }
  {
    unsigned int* z = reinterpret_cast<unsigned int*>(&s_attnh[0][0][0]);
    for (int i = tid; i < (8 * 16 * 32) / 2; i += 256) z[i] = 0u;
  }
  {
    const _Float16* vbase = qkv + (size_t)(seqB * 16) * C3 + 2 * CC + hd * CHD;
    for (int r = 0; r < 16; ++r) {
      unsigned int pv =
          *reinterpret_cast<const unsigned int*>(vbase + (size_t)r * C3 + 2 * lane);
      _Float16 h2[2];
      __builtin_memcpy(h2, &pv, 4);
      s_vT[w][2 * lane + 0][r] = h2[0];
      s_vT[w][2 * lane + 1][r] = h2[1];
    }
  }
  __syncthreads();

  // --- content scores: S = (q*s)(k*s)^T, K = 64 -> two WMMAs ---
  const _Float16* qbase = qkv + (size_t)(seqB * 16) * C3 + hd * CHD;
  const _Float16* kbase = qkv + (size_t)(seqB * 16) * C3 + CC + hd * CHD;
  v8f sc = {};
  sc = wmma_f16(load_frag(qbase, C3),      load_frag(kbase, C3),      sc);
  sc = wmma_f16(load_frag(qbase + 32, C3), load_frag(kbase + 32, C3), sc);
#pragma unroll
  for (int v = 0; v < 8; ++v) s_S[w][v + 8 * half][n] = sc[v];
  __syncthreads();

  // --- rel-pos key term: S[q][k] += sum_c qs[q][c] * kt[k-q+15][c] ---
  {
    const int q  = n;
    const int ks = half * 8;
    const _Float16* qrow = qbase + (size_t)q * C3;
    float acc2[8] = {0, 0, 0, 0, 0, 0, 0, 0};
    const int dbase = ks - q + 15;
    for (int c = 0; c < 64; ++c) {
      float qc = (float)qrow[c];
#pragma unroll
      for (int kk = 0; kk < 8; ++kk) acc2[kk] += qc * s_kt[dbase + kk][c];
    }
#pragma unroll
    for (int kk = 0; kk < 8; ++kk) s_S[w][q][ks + kk] += acc2[kk];
  }
  __syncthreads();

  // --- softmax over k (lanes 0..15 each own a q row) ---
  if (lane < 16) {
    const int q = lane;
    float mx = -3.4e38f;
#pragma unroll
    for (int k = 0; k < 16; ++k) mx = fmaxf(mx, s_S[w][q][k]);
    float e[16]; float sum = 0.f;
#pragma unroll
    for (int k = 0; k < 16; ++k) { e[k] = __expf(s_S[w][q][k] - mx); sum += e[k]; }
    const float inv = 1.0f / sum;
#pragma unroll
    for (int k = 0; k < 16; ++k) {
      float a = e[k] * inv;
      s_S[w][q][k]     = a;             // f32 copy for rel-pos V term
      s_attnh[w][q][k] = (_Float16)a;   // f16 for WMMA (cols 16..31 stay 0)
    }
  }
  __syncthreads();

  // --- AV: a[q][c] = attn @ v  (4 col tiles, K=16 padded) + rel-pos V ---
  v16h afrag = load_frag(&s_attnh[0][0][0] + (size_t)w * 16 * 32, 32);
#pragma unroll
  for (int tc = 0; tc < 4; ++tc) {
    const int cc = tc * 16 + n;
    v16h bfrag = load_frag_k16(&s_vT[w][cc][0]);
    v8f av = {};
    av = wmma_f16(afrag, bfrag, av);
#pragma unroll
    for (int v = 0; v < 8; ++v) {
      const int q = v + 8 * half;
      float rp = 0.f;
#pragma unroll
      for (int k = 0; k < 16; ++k) rp += s_S[w][q][k] * s_vt[k - q + 15][cc];
      float val = av[v] + rp;
      aout[(size_t)(seqB * 16 + q) * CC + hd * CHD + cc] = (_Float16)val;
    }
  }
}

// ---------------------------------------------------------------------------
// K4: out = x + a @ Wp^T + pb. Weight tile (16 cols x 512 K, 16 KB) staged
//     via async->LDS; results restaged through LDS so global writes are
//     contiguous over wi (32-float runs matching the (b,c,t,h,w) layout).
// ---------------------------------------------------------------------------
__global__ void k_proj_gemm(const _Float16* __restrict__ a,
                            const _Float16* __restrict__ wp,
                            const float* __restrict__ pb,
                            const float* __restrict__ x,
                            float* __restrict__ out) {
  __shared__ float    s_out[512][17];    // padded to dodge bank conflicts
  __shared__ _Float16 s_b[16 * CC];      // 16 weight rows x K
  const int tid  = threadIdx.x;
  const int w    = tid >> 5;
  const int lane = tid & 31;
  const int n    = lane & 15;
  const int half = lane >> 4;
  const int j0 = blockIdx.x * 16;        // 32 col tiles
  const int R0 = blockIdx.y * 512;       // 64 row groups (32 seqB each)
  const int rbase = R0 + w * 64;         // wave: 4 row tiles

  // async stage of the block's weight tile into LDS
  {
    const _Float16* btile = wp + (size_t)j0 * CC;
#pragma unroll
    for (int it = 0; it < 4; ++it) {
      int i = (it * 256 + tid) * 8;      // 1024 x b128 chunks
      async_copy_b128(btile + i, &s_b[i]);
    }
    wait_async_lds();
  }
  __syncthreads();

  v8f acc[4];
#pragma unroll
  for (int i = 0; i < 4; ++i) acc[i] = (v8f){};

  for (int kk = 0; kk < 16; ++kk) {
    v16h bf = load_frag(&s_b[kk * 32], CC);     // LDS
#pragma unroll
    for (int i = 0; i < 4; ++i) {
      v16h af = load_frag(a + (size_t)(rbase + i * 16) * CC + kk * 32, CC);
      acc[i] = wmma_f16(af, bf, acc[i]);
    }
  }
#pragma unroll
  for (int i = 0; i < 4; ++i)
#pragma unroll
    for (int v = 0; v < 8; ++v)
      s_out[w * 64 + i * 16 + v + 8 * half][n] = acc[i][v];
  __syncthreads();

  const int bb = blockIdx.y >> 5;
  const int hi = blockIdx.y & 31;
  const int wi = tid & 31;
  const int g  = tid >> 5;
  for (int it = 0; it < 32; ++it) {
    const int idx2 = it * 8 + g;           // 0..255 -> (jp, tq)
    const int jp = idx2 >> 4;
    const int tq = idx2 & 15;
    float val = s_out[wi * 16 + tq][jp] + pb[j0 + jp];
    size_t addr = ((size_t)(bb * CC + j0 + jp) * CT + tq) * 1024 +
                  (size_t)hi * 32 + wi;
    out[addr] = x[addr] + val;
  }
}

// ---------------------------------------------------------------------------
extern "C" void kernel_launch(void* const* d_in, const int* in_sizes, int n_in,
                              void* d_out, int out_size, void* d_ws,
                              size_t ws_size, hipStream_t stream) {
  (void)in_sizes; (void)n_in; (void)out_size; (void)ws_size;
  const float* x      = (const float*)d_in[0];
  const float* ln_g   = (const float*)d_in[1];
  const float* ln_b   = (const float*)d_in[2];
  const float* qkv_w  = (const float*)d_in[3];
  const float* qkv_b  = (const float*)d_in[4];
  const float* ktab   = (const float*)d_in[5];
  const float* vtab   = (const float*)d_in[6];
  const float* proj_w = (const float*)d_in[7];
  const float* proj_b = (const float*)d_in[8];
  float* out = (float*)d_out;

  char* ws = (char*)d_ws;
  _Float16* wsQKV = (_Float16*)(ws);                    // 32768*1536*2 = 100663296
  _Float16* wsA   = (_Float16*)(ws + 100663296);        // 32768*512*2  =  33554432
  _Float16* wsWq  = (_Float16*)(ws + 134217728);        // 1536*512*2   =   1572864
  _Float16* wsWp  = (_Float16*)(ws + 135790592);        // 512*512*2    =    524288

  k_f32_to_f16<<<(C3 * CC + 255) / 256, 256, 0, stream>>>(qkv_w, wsWq, C3 * CC);
  k_f32_to_f16<<<(CC * CC + 255) / 256, 256, 0, stream>>>(proj_w, wsWp, CC * CC);

  // wsA first holds the LayerNorm output (nrm), later the attention output.
  k_layernorm<<<CB * CT * CHH, dim3(32, 8), 0, stream>>>(x, ln_g, ln_b, wsA);
  k_qkv_gemm<<<dim3(C3 / 32, ROWS / 64), 256, 0, stream>>>(wsA, wsWq, qkv_b, wsQKV);
  k_attention<<<SEQ, 256, 0, stream>>>(wsQKV, ktab, vtab, wsA);
  k_proj_gemm<<<dim3(CC / 16, ROWS / 512), 256, 0, stream>>>(wsA, wsWp, proj_b, x, out);
}
